// PagedGroupedQueryAttention_70677981822985
// MI455X (gfx1250) — compile-verified, matrix operand order checked by reading, not run on hardware
//
#include <hip/hip_runtime.h>
#include <hip/hip_bf16.h>

// ---------------- problem constants ----------------
#define PB   4      // batch
#define PT   2048   // seq len
#define PD   2048   // model dim
#define PH   16     // query heads
#define PG   4      // kv heads
#define PHD  128    // head dim
#define PNB  512    // num paged blocks
#define PBS  16     // page (block) size
#define PSCALE 0.08838834764831845f   // 1/sqrt(128)
#define PNEG  (-1e30f)

#if defined(__gfx1250__) && __has_builtin(__builtin_amdgcn_tensor_load_to_lds)
#define PHAS_TDM 1
#else
#define PHAS_TDM 0
#endif

typedef __bf16 v16bf __attribute__((ext_vector_type(16)));
typedef float  v8f   __attribute__((ext_vector_type(8)));
typedef unsigned int u32x4 __attribute__((ext_vector_type(4)));
typedef int          i32x8 __attribute__((ext_vector_type(8)));
typedef int          i32x4 __attribute__((ext_vector_type(4)));

struct __align__(16) U4 { unsigned int a, b, c, d; };
struct __align__(8)  B4 { __bf16 e[4]; };
struct __align__(16) B8 { __bf16 e[8]; };
union BFrag { v16bf v; U4 q[2]; };

__device__ __forceinline__ __bf16 f2bf(float f) {
  union { float f; unsigned u; } a; a.f = f;
  unsigned r = (a.u + 0x7FFFu + ((a.u >> 16) & 1u)) >> 16;
  union { unsigned short s; __bf16 b; } o; o.s = (unsigned short)r;
  return o.b;
}

// vectorized load-4 + convert helpers (f32 source uses b128 loads)
__device__ __forceinline__ B4 load4cvt(const float* p) {
  const float4 v = *(const float4*)p;
  B4 d; d.e[0] = f2bf(v.x); d.e[1] = f2bf(v.y); d.e[2] = f2bf(v.z); d.e[3] = f2bf(v.w);
  return d;
}
__device__ __forceinline__ B4 load4cvt(const __bf16* p) { return *(const B4*)p; }

__device__ __forceinline__ v8f vzero8() {
  v8f v;
  #pragma unroll
  for (int i = 0; i < 8; ++i) v[i] = 0.f;
  return v;
}

// Load one 16x32-bf16 A/B fragment slice for this lane.
// Per CDNA5 ISA 7.12.2 (16-bit 16x32): element j -> k = (j&7) + (j>>3)*16 + (lane>=16)*8.
// Caller passes base = &row[(lane>>4)*8]; we fetch k..k+7 and k+16..k+23 (two b128 loads).
__device__ __forceinline__ v16bf load_frag(const __bf16* base) {
  BFrag f;
  f.q[0] = *(const U4*)(base);
  f.q[1] = *(const U4*)(base + 16);
  return f.v;
}

__device__ __forceinline__ v8f wmma_bf16(v16bf a, v16bf b, v8f c) {
  // v_wmma_f32_16x16x32_bf16: (neg_a, A, neg_b, B, c_mod, C, reuse_a, reuse_b)
  return __builtin_amdgcn_wmma_f32_16x16x32_bf16(false, a, false, b, (short)0, c,
                                                 false, false);
}

#if PHAS_TDM
// Issue one TDM copy of a contiguous run of `nelem` bf16 (one KV page = 2048)
// from global memory into LDS at byte offset lds_off. D# per CDNA5 ISA §8.
// This toolchain exposes the 6-arg builtin: (g0, g1, g2, g3, g4, cpol).
__device__ __forceinline__ void tdm_load_run(unsigned lds_off, const __bf16* gsrc,
                                             unsigned nelem) {
  unsigned long long ga = (unsigned long long)gsrc;
  u32x4 g0;
  g0[0] = 1u;                                   // count=1, user mode
  g0[1] = lds_off;                              // lds_addr (bytes)
  g0[2] = (unsigned)(ga & 0xFFFFFFFFu);         // global_addr[31:0]
  g0[3] = (unsigned)((ga >> 32) & 0x1FFFFFFu)   // global_addr[56:32]
        | (2u << 30);                           // type = 2 ("image")
  i32x8 g1;
  g1[0] = 1 << 16;                              // workgroup_mask=0, data_size=1 (2B)
  g1[1] = (int)(nelem << 16);                   // tensor_dim0[15:0] -> bits[63:48]
  g1[2] = 1 << 16;                              // tensor_dim0 hi=0; tensor_dim1=1
  g1[3] = (int)(nelem << 16);                   // tensor_dim1 hi=0; tile_dim0=nelem
  g1[4] = 1;                                    // tile_dim1=1, tile_dim2=0
  g1[5] = (int)nelem;                           // tensor_dim0_stride lo
  g1[6] = 0;
  g1[7] = 0;
  i32x4 z4; z4[0] = 0; z4[1] = 0; z4[2] = 0; z4[3] = 0;  // groups 2/3 unused (<=2D)
  i32x8 z8;
  #pragma unroll
  for (int i = 0; i < 8; ++i) z8[i] = 0;
  __builtin_amdgcn_tensor_load_to_lds(g0, g1, z4, z4, z8, 0);
}
#endif

// ---------------- elementwise f32 -> bf16 ----------------
__global__ __launch_bounds__(256) void convert_kernel(const float* __restrict__ in,
                                                      __bf16* __restrict__ out, int n) {
  int i = blockIdx.x * 256 + threadIdx.x;
  if (i < n) out[i] = f2bf(in[i]);
}

// ---------------- WMMA GEMM: C[M,N] = A[M,K] * B[K,N] ----------------
// MODE 0: bf16 row-major; MODE 1: bf16 scattered to [B,H,T,HD]; MODE 2: f32 row-major.
template <int MODE, typename AT>
__global__ __launch_bounds__(256) void gemm_kernel(const AT* __restrict__ A,
                                                   const float* __restrict__ Bw,
                                                   void* __restrict__ Cout,
                                                   int M, int N, int K) {
  __shared__ __align__(16) __bf16 As[128 * 32];   // [m][k]
  __shared__ __align__(16) __bf16 Bs[128 * 32];   // transposed: [n][k]

  const int tid  = threadIdx.x;
  const int wave = tid >> 5;
  const int lane = tid & 31;
  const int lhi  = lane >> 4;      // 0/1 half-wave
  const int llo  = lane & 15;
  const int wm   = wave & 3;       // 4 wave rows  * 32
  const int wn   = wave >> 2;      // 2 wave cols  * 64
  const int m0   = blockIdx.y * 128;
  const int n0   = blockIdx.x * 128;

  v8f acc[2][4];
  #pragma unroll
  for (int s = 0; s < 2; ++s)
    #pragma unroll
    for (int c = 0; c < 4; ++c) acc[s][c] = vzero8();

  for (int k0 = 0; k0 < K; k0 += 32) {
    __syncthreads();
    // Stage A tile (128x32): vector loads, packed 4x-bf16 (b64) LDS stores.
    #pragma unroll
    for (int i = tid; i < 128 * 8; i += 256) {
      int r = i >> 3, c4 = (i & 7) * 4;
      *(B4*)(As + r * 32 + c4) = load4cvt(&A[(size_t)(m0 + r) * K + (k0 + c4)]);
      if (k0 + 32 < K)
        __builtin_prefetch(&A[(size_t)(m0 + r) * K + (k0 + 32 + c4)], 0, 1);
    }
    // Stage B tile transposed (Bs[n][k]): vector global reads, scatter LDS writes.
    #pragma unroll
    for (int i = tid; i < 32 * 32; i += 256) {
      int kk = i >> 5, n4 = (i & 31) * 4;
      B4 v = load4cvt(&Bw[(size_t)(k0 + kk) * N + (n0 + n4)]);
      #pragma unroll
      for (int e = 0; e < 4; ++e) Bs[(n4 + e) * 32 + kk] = v.e[e];
    }
    __syncthreads();

    v16bf af[2];
    #pragma unroll
    for (int s = 0; s < 2; ++s)
      af[s] = load_frag(As + (wm * 32 + s * 16 + llo) * 32 + lhi * 8);
    #pragma unroll
    for (int c = 0; c < 4; ++c) {
      v16bf bf = load_frag(Bs + (wn * 64 + c * 16 + llo) * 32 + lhi * 8);
      #pragma unroll
      for (int s = 0; s < 2; ++s) acc[s][c] = wmma_bf16(af[s], bf, acc[s][c]);
    }
  }

  // Epilogue. C fragment layout: VGPR r -> row m = r + 8*(lane>=16), col n = lane&15.
  #pragma unroll
  for (int s = 0; s < 2; ++s)
    #pragma unroll
    for (int c = 0; c < 4; ++c)
      #pragma unroll
      for (int r = 0; r < 8; ++r) {
        int m = m0 + wm * 32 + s * 16 + r + 8 * lhi;
        int n = n0 + wn * 64 + c * 16 + llo;
        float v = acc[s][c][r];
        if (MODE == 2) {
          ((float*)Cout)[(size_t)m * N + n] = v;
        } else if (MODE == 0) {
          ((__bf16*)Cout)[(size_t)m * N + n] = f2bf(v);
        } else {  // q layout [B,H,T,HD]
          int b = m >> 11, t = m & (PT - 1);
          int h = n >> 7,  hd = n & (PHD - 1);
          ((__bf16*)Cout)[(((size_t)(b * PH + h) * PT + t) << 7) + hd] = f2bf(v);
        }
      }
}

// ---------------- paged KV scatter (honors block_tables / kv_lens) ----------------
__global__ __launch_bounds__(256) void scatter_kv_kernel(const __bf16* __restrict__ Ck,
                                                         const __bf16* __restrict__ Cv,
                                                         const int* __restrict__ block_tables,
                                                         const int* __restrict__ kv_lens,
                                                         __bf16* __restrict__ kcb,
                                                         __bf16* __restrict__ vcb) {
  int i = blockIdx.x * 256 + threadIdx.x;
  if (i >= PB * PT * PG * PHD) return;
  int hd = i & (PHD - 1);
  int g  = (i >> 7) & (PG - 1);
  int t  = (i >> 9) & (PT - 1);
  int b  = i >> 20;
  int pos  = kv_lens[b] + t;
  int bid  = block_tables[b * (PT / PBS) + (pos >> 4)];
  size_t dst = (((size_t)(bid * PG + g) * PBS) + (pos & 15)) * PHD + hd;
  size_t src = ((size_t)(b * PT) + t) * (PG * PHD) + g * PHD + hd;
  kcb[dst] = Ck[src];
  vcb[dst] = Cv[src];
}

// ---------------- flash attention over paged KV ----------------
// LDS carved from one static block so TDM lds_addr offsets are well-defined.
#define LDS_KS_OFF   0u                       // Ks[32][128] bf16   (8 KB)
#define LDS_VT_OFF   8192u                    // Vt[128][32] bf16   (8 KB)
#define LDS_PS_OFF   16384u                   // Pscr[8][16*32] bf16 (8 KB)
__global__ __launch_bounds__(256) void attn_kernel(const __bf16* __restrict__ qb,
                                                   const __bf16* __restrict__ kcb,
                                                   const __bf16* __restrict__ vcb,
                                                   const int* __restrict__ block_tables,
                                                   __bf16* __restrict__ attn) {
  __shared__ __align__(128) char smem[24576];
  __bf16* Ks   = (__bf16*)(smem + LDS_KS_OFF);   // [key][hd]
  __bf16* Vt   = (__bf16*)(smem + LDS_VT_OFF);   // [hd][key]
  __bf16* Pscr = (__bf16*)(smem + LDS_PS_OFF);   // per-wave P scratch

  const int b    = blockIdx.z;
  const int h    = blockIdx.y;
  const int g    = h & (PG - 1);                 // h = j*G + g
  const int tid  = threadIdx.x;
  const int wave = tid >> 5;
  const int lane = tid & 31;
  const int lhi  = lane >> 4;
  const int llo  = lane & 15;
  const int qt0  = blockIdx.x * 128;
  const int qm0  = qt0 + wave * 16;              // this wave's 16 query rows

  // Q fragments (16 rows x 128 dims = 4 K-chunks), straight from global (bf16).
  v16bf qf[4];
  {
    const __bf16* qrow = qb + ((size_t)(b * PH + h) * PT + (qm0 + llo)) * PHD;
    #pragma unroll
    for (int c = 0; c < 4; ++c) qf[c] = load_frag(qrow + c * 32 + lhi * 8);
  }

  v8f o[8];
  #pragma unroll
  for (int c = 0; c < 8; ++c) o[c] = vzero8();
  float mst[8], lst[8];
  #pragma unroll
  for (int r = 0; r < 8; ++r) { mst[r] = PNEG; lst[r] = 0.f; }

  const int send = qt0 + 128;  // causal bound for the whole block
  for (int s0 = 0; s0 < send; s0 += 32) {
    __syncthreads();  // previous tile fully consumed before restage

#if PHAS_TDM
    // K tile: two contiguous 4KB pages moved by the Tensor Data Mover.
    if (wave == 0) {
      int bid0 = block_tables[b * (PT / PBS) + (s0 >> 4)];
      int bid1 = block_tables[b * (PT / PBS) + (s0 >> 4) + 1];
      tdm_load_run(LDS_KS_OFF,
                   kcb + (size_t)(bid0 * PG + g) * (PBS * PHD), PBS * PHD);
      tdm_load_run(LDS_KS_OFF + PBS * PHD * 2,
                   kcb + (size_t)(bid1 * PG + g) * (PBS * PHD), PBS * PHD);
      __builtin_amdgcn_s_wait_tensorcnt(0);
    }
#endif
    // V^T tile (transpose scatter) — and K tile fallback when TDM is absent.
    #pragma unroll
    for (int i = tid; i < 512; i += 256) {      // 512 chunks of 8 bf16
      int kl = i >> 4;                          // key 0..31
      int hb = (i & 15) * 8;                    // hd base
      int s  = s0 + kl;
      int bid = block_tables[b * (PT / PBS) + (s >> 4)];
      size_t idx = (((size_t)(bid * PG + g) * PBS) + (s & 15)) * PHD + hb;
      B8 v = *(const B8*)(vcb + idx);
      #pragma unroll
      for (int e = 0; e < 8; ++e) Vt[(hb + e) * 32 + kl] = v.e[e];
#if !PHAS_TDM
      *(B8*)(Ks + kl * PHD + hb) = *(const B8*)(kcb + idx);
#endif
    }
    __syncthreads();  // publish Ks/Vt to all waves

    if (s0 <= qm0 + 15) {  // tile intersects causal region for this wave
      float st[2][8];
      #pragma unroll
      for (int sub = 0; sub < 2; ++sub) {
        v8f sacc = vzero8();
        const __bf16* krow = Ks + (sub * 16 + llo) * PHD;  // lane's key row
        #pragma unroll
        for (int c = 0; c < 4; ++c)
          sacc = wmma_bf16(qf[c], load_frag(krow + c * 32 + lhi * 8), sacc);
        #pragma unroll
        for (int r = 0; r < 8; ++r) {
          int t = qm0 + r + 8 * lhi;
          int s = s0 + sub * 16 + llo;
          float f = sacc[r] * PSCALE;
          st[sub][r] = (s > t) ? PNEG : f;
        }
      }
      // online softmax: row reductions across the 16 lanes of each half-wave
      float alpha[8];
      #pragma unroll
      for (int r = 0; r < 8; ++r) {
        float v = fmaxf(st[0][r], st[1][r]);
        #pragma unroll
        for (int d = 1; d < 16; d <<= 1) v = fmaxf(v, __shfl_xor(v, d, 32));
        float mnew = fmaxf(mst[r], v);
        alpha[r] = __expf(mst[r] - mnew);
        float p0 = __expf(st[0][r] - mnew);
        float p1 = __expf(st[1][r] - mnew);
        st[0][r] = p0; st[1][r] = p1;
        float rs = p0 + p1;
        #pragma unroll
        for (int d = 1; d < 16; d <<= 1) rs += __shfl_xor(rs, d, 32);
        lst[r] = lst[r] * alpha[r] + rs;
        mst[r] = mnew;
      }
      #pragma unroll
      for (int c = 0; c < 8; ++c)
        #pragma unroll
        for (int r = 0; r < 8; ++r) o[c][r] *= alpha[r];

      // C-layout P -> A-layout via per-wave LDS scratch (same-wave DS order).
      __bf16* ps = Pscr + wave * (16 * 32);
      #pragma unroll
      for (int sub = 0; sub < 2; ++sub)
        #pragma unroll
        for (int r = 0; r < 8; ++r)
          ps[(r + 8 * lhi) * 32 + sub * 16 + llo] = f2bf(st[sub][r]);
      v16bf pf = load_frag(ps + llo * 32 + lhi * 8);

      // P(16x32) @ V(32x128): 8 n-chunks of 16.
      #pragma unroll
      for (int c = 0; c < 8; ++c) {
        v16bf vf = load_frag(Vt + (c * 16 + llo) * 32 + lhi * 8);
        o[c] = wmma_bf16(pf, vf, o[c]);
      }
    }
  }

  // normalize and write attn output row-major [B*T, H*HD] (bf16 for final GEMM)
  float inv[8];
  #pragma unroll
  for (int r = 0; r < 8; ++r) inv[r] = 1.f / lst[r];
  #pragma unroll
  for (int c = 0; c < 8; ++c)
    #pragma unroll
    for (int r = 0; r < 8; ++r) {
      int t  = qm0 + r + 8 * lhi;
      int hd = c * 16 + llo;
      attn[((size_t)(b * PT) + t) * (PH * PHD) + h * PHD + hd] = f2bf(o[c][r] * inv[r]);
    }
}

// ---------------- host-side launch ----------------
extern "C" void kernel_launch(void* const* d_in, const int* in_sizes, int n_in,
                              void* d_out, int out_size, void* d_ws, size_t ws_size,
                              hipStream_t stream) {
  const float* x    = (const float*)d_in[0];
  const float* w_q  = (const float*)d_in[1];
  const float* w_k  = (const float*)d_in[2];
  const float* w_v  = (const float*)d_in[3];
  const float* w_o  = (const float*)d_in[4];
  const float* kbl  = (const float*)d_in[5];
  const float* vbl  = (const float*)d_in[6];
  const int* btab   = (const int*)d_in[7];
  const int* klen   = (const int*)d_in[8];
  float* out        = (float*)d_out;

  const size_t SZ_QB  = (size_t)PB * PH * PT * PHD * 2;   // 32 MB
  const size_t SZ_CKV = (size_t)PB * PT * PG * PHD * 2;   // 8 MB
  const size_t SZ_KCB = (size_t)PNB * PG * PBS * PHD * 2; // 8 MB

  char* w = (char*)d_ws;
  __bf16* qb   = (__bf16*)w;                      w += SZ_QB;
  __bf16* Ck   = (__bf16*)w;                      w += SZ_CKV;
  __bf16* Cv   = (__bf16*)w;                      w += SZ_CKV;
  __bf16* kcb  = (__bf16*)w;                      w += SZ_KCB;
  __bf16* vcb  = (__bf16*)w;                      w += SZ_KCB;
  __bf16* attn = (__bf16*)w;                      w += SZ_QB;
  (void)ws_size; (void)in_sizes; (void)n_in; (void)out_size;

  const int M = PB * PT;        // 8192
  const int nKV = PNB * PG * PBS * PHD;  // 4,194,304

  // init paged KV workspace from input blocks (bf16)
  convert_kernel<<<(nKV + 255) / 256, 256, 0, stream>>>(kbl, kcb, nKV);
  convert_kernel<<<(nKV + 255) / 256, 256, 0, stream>>>(vbl, vcb, nKV);

  // QKV projections (f32 inputs converted to bf16 while staging into LDS)
  gemm_kernel<1, float><<<dim3(PD / 128, M / 128), 256, 0, stream>>>(x, w_q, qb, M, PH * PHD, PD);
  gemm_kernel<0, float><<<dim3((PG * PHD) / 128, M / 128), 256, 0, stream>>>(x, w_k, Ck, M, PG * PHD, PD);
  gemm_kernel<0, float><<<dim3((PG * PHD) / 128, M / 128), 256, 0, stream>>>(x, w_v, Cv, M, PG * PHD, PD);

  // paged scatter of new K/V into block pool
  int nScat = PB * PT * PG * PHD;
  scatter_kv_kernel<<<(nScat + 255) / 256, 256, 0, stream>>>(Ck, Cv, btab, klen, kcb, vcb);

  // flash attention over paged KV (K pages staged via Tensor Data Mover)
  attn_kernel<<<dim3(PT / 128, PH, PB), 256, 0, stream>>>(qb, kcb, vcb, btab, attn);

  // output projection -> f32 result
  gemm_kernel<2, __bf16><<<dim3(PD / 128, M / 128), 256, 0, stream>>>(attn, w_o, out, M, PD, PH * PHD);
}
